// MonarchAttention_31774168056238
// MI455X (gfx1250) — matrix operand order repaired and checked by reading, not compiled, not run on hardware
//
#include <hip/hip_runtime.h>
#include <hip/hip_bf16.h>

// ---------------------------------------------------------------------------
// MonarchAttention for MI455X (gfx1250), wave32 + WMMA bf16 (f32 accumulate).
// Problem: B=4,H=16 (BH=64), SEQ=2000, HD=64, BLOCK=64, NB=32, PAD=48, 3 steps.
// Bandwidth-bound (~21 GFLOP vs ~400MB moved): bf16 intermediates, LDS tiling,
// all GEMMs on v_wmma_f32_16x16x32_bf16, async global->LDS for bf16 tiles.
// ---------------------------------------------------------------------------

typedef __bf16 bf16_t;
typedef __attribute__((ext_vector_type(16))) __bf16 v16bf;
typedef __attribute__((ext_vector_type(8)))  float  v8f;
typedef int v4i __attribute__((__vector_size__(16)));   // builtin's pointee type

#define BHN 64          // B*H
#define NB  32          // NUM_BLOCKS
#define BS  64          // BLOCK_SIZE
#define HD  64          // HEAD_DIM
#define SEQ 2000
#define PADN 48
#define MASK_I0 (BS - PADN)            // 16
#define INV_N2 (1.0f / (2000.0f * 2000.0f))

#if defined(__has_builtin)
#if __has_builtin(__builtin_amdgcn_global_load_async_to_lds_b128)
#define HAVE_ASYNC_LDS 1
#endif
#endif
#ifndef HAVE_ASYNC_LDS
#define HAVE_ASYNC_LDS 0
#endif

__device__ __forceinline__ void wait_asynccnt0() {
#if defined(__has_builtin) && __has_builtin(__builtin_amdgcn_s_wait_asynccnt)
  __builtin_amdgcn_s_wait_asynccnt(0);
#else
  asm volatile("s_wait_asynccnt 0x0" ::: "memory");
#endif
}

// Copy nrows rows of rowElems bf16 (rowElems % 8 == 0, 16B aligned) from global
// (row stride gstride elems) into contiguous LDS rows. 256-thread cooperative.
__device__ __forceinline__ void copy_rows_bf16(bf16_t* __restrict__ lds,
                                               const bf16_t* __restrict__ g,
                                               int nrows, int rowElems,
                                               size_t gstride) {
  int t = threadIdx.x;
  int gpr = rowElems >> 3;             // 16B granules per row
  int total = nrows * gpr;
#if HAVE_ASYNC_LDS
  for (int idx = t; idx < total; idx += 256) {
    int r = idx / gpr, c = (idx - r * gpr) << 3;
    const bf16_t* gp = g + (size_t)r * gstride + c;
    bf16_t* lp = lds + (size_t)r * rowElems + c;
    __builtin_amdgcn_global_load_async_to_lds_b128((v4i*)gp, (v4i*)lp, 0, 0);
  }
  wait_asynccnt0();
#else
  for (int idx = t; idx < total; idx += 256) {
    int r = idx / gpr, c = (idx - r * gpr) << 3;
    *(v4i*)(lds + (size_t)r * rowElems + c) =
        *(const v4i*)(g + (size_t)r * gstride + c);
  }
#endif
}

__device__ __forceinline__ v8f wmma_bf16(v16bf a, v16bf b, v8f c) {
  // 8 args: (neg_a, A, neg_b, B, c_mod, C, reuse_a, reuse_b)
  return __builtin_amdgcn_wmma_f32_16x16x32_bf16(false, a, false, b, (short)0, c,
                                                 false, false);
}

// A fragment (16x32 bf16, M x K) from row-major [M][K] LDS, leading dim ld.
// Layout: lanes 0-15 row M, elems K0..15; lanes 16-31 K16..31 (elem e -> K=kb+e).
__device__ __forceinline__ v16bf frag_A(const bf16_t* src, int ld, int m0, int k0) {
  int lane = threadIdx.x & 31;
  int m  = m0 + (lane & 15);
  int kb = k0 + ((lane >> 4) << 4);
  v16bf a;
#pragma unroll
  for (int e = 0; e < 16; ++e) a[e] = src[m * ld + kb + e];
  return a;
}

// B fragment from row-major [N][K] LDS (i.e. computes A * Y^T) - contiguous reads.
// Lanes 0-15: col N, K=0..15; lanes 16-31: col N, K=16..31 (elem e -> K=kb+e).
__device__ __forceinline__ v16bf frag_B_nk(const bf16_t* src, int ld, int k0, int n0) {
  int lane = threadIdx.x & 31;
  int n  = n0 + (lane & 15);
  int kb = k0 + ((lane >> 4) << 4);
  v16bf b;
#pragma unroll
  for (int e = 0; e < 16; ++e) b[e] = src[n * ld + kb + e];
  return b;
}

// C/D layout: VGPR r, lane: row = m0 + (lane>=16 ? 8:0) + r, col = n0 + (lane&15).
__device__ __forceinline__ void store_D_f32(float* dst, int ld, int m0, int n0, v8f d) {
  int lane = threadIdx.x & 31;
  int n  = n0 + (lane & 15);
  int mb = m0 + ((lane >> 4) << 3);
#pragma unroll
  for (int r = 0; r < 8; ++r) dst[(mb + r) * ld + n] = d[r];
}
__device__ __forceinline__ void store_D_bf16(bf16_t* dst, int ld, int m0, int n0, v8f d) {
  int lane = threadIdx.x & 31;
  int n  = n0 + (lane & 15);
  int mb = m0 + ((lane >> 4) << 3);
#pragma unroll
  for (int r = 0; r < 8; ++r) dst[(size_t)(mb + r) * ld + n] = (bf16_t)d[r];
}

// ---------------------------------------------------------------------------
// S1: per (bh,k): mask+normalize right rows, r2[k,j], T[k,j,a] = right @ Kblk
// ---------------------------------------------------------------------------
__global__ __launch_bounds__(256)
void k_step_right_T(const float* __restrict__ Kg, const float* __restrict__ Rws,
                    bf16_t* __restrict__ Tws, float* __restrict__ r2ws) {
  int gid = blockIdx.x;
  int bh = gid / NB, k = gid % NB;
  __shared__ float  sRow[BS][BS + 1];   // fp32 masked R staging (+1: bank spread)
  __shared__ bf16_t sRight[BS][BS];     // right = normalize(R)^2   [j][i]
  __shared__ bf16_t sKT[HD][BS];        // K block transposed       [a][i]
  int t = threadIdx.x;
  const float* Rk = Rws + ((size_t)bh * NB + k) * BS * BS;
  for (int idx = t; idx < BS * BS; idx += 256) {
    int j = idx >> 6, i = idx & 63;
    float v = Rk[idx];
    if (k == NB - 1 && i >= MASK_I0) v = 0.0f;
    sRow[j][i] = v;
  }
  const float* Kbh = Kg + (size_t)bh * SEQ * HD;
  for (int idx = t; idx < BS * HD; idx += 256) {
    int i = idx >> 6, a = idx & 63;
    int s = k * BS + i;
    sKT[a][i] = (bf16_t)((s < SEQ) ? Kbh[(size_t)s * HD + a] : 0.0f);
  }
  __syncthreads();
  if (t < BS) {
    int j = t;
    float n2 = 0.0f;
    for (int i = 0; i < BS; ++i) { float v = sRow[j][i]; n2 += v * v; }
    float inv = 1.0f / fmaxf(sqrtf(n2), 1e-12f);
    float r2 = 0.0f;
    for (int i = 0; i < BS; ++i) {
      float rs = sRow[j][i] * inv;
      float rt = rs * rs;
      r2 += rt * rt;
      sRight[j][i] = (bf16_t)rt;
    }
    r2ws[((size_t)bh * NB + k) * BS + j] = r2;
  }
  __syncthreads();
  int wave = t >> 5;
  bf16_t* Tk = Tws + ((size_t)bh * NB + k) * BS * HD;
  for (int tile = wave; tile < 16; tile += 8) {   // 64x64 out, 4x4 tiles
    int tm = (tile >> 2) << 4, tn = (tile & 3) << 4;
    v8f acc = {};
    acc = wmma_bf16(frag_A(&sRight[0][0], BS, tm, 0),  frag_B_nk(&sKT[0][0], BS, 0,  tn), acc);
    acc = wmma_bf16(frag_A(&sRight[0][0], BS, tm, 32), frag_B_nk(&sKT[0][0], BS, 32, tn), acc);
    store_D_bf16(Tk, HD, tm, tn, acc);
  }
}

// ---------------------------------------------------------------------------
// S2: per (bh,j): qk_l = Q_j @ T_j^T; d_left + update L; l2; U = left^T @ Q_j
// ---------------------------------------------------------------------------
__global__ __launch_bounds__(256)
void k_step_left(const float* __restrict__ Qg, const bf16_t* __restrict__ Tws,
                 const float* __restrict__ r2ws, float* __restrict__ Lws,
                 bf16_t* __restrict__ Uws, float* __restrict__ l2ws) {
  int gid = blockIdx.x;
  int bh = gid / BS, j = gid % BS;
  __shared__ bf16_t sQ[NB][HD];          // [l][a]
  __shared__ bf16_t sQT[HD][NB];         // [a][l]  (transposed copy for U GEMM)
  __shared__ bf16_t sT[NB][HD];          // [k][a]
  __shared__ float  sQKl[NB][NB + 1];    // [l][k]
  __shared__ float  sL[NB][NB];          // original L_j
  __shared__ float  sLS[NB][NB];         // ls
  __shared__ bf16_t sLeftT[NB][NB];      // left^T [k][l] (bf16 A-operand)
  __shared__ float  sR2[NB];
  int t = threadIdx.x;
  // async copy: T_j rows [k][a], global row stride BS*HD, pure bf16 b128 copy
  const bf16_t* Tbh = Tws + (size_t)bh * NB * BS * HD;
  copy_rows_bf16(&sT[0][0], Tbh + (size_t)j * HD, NB, HD, (size_t)BS * HD);
  const float* Qbh = Qg + (size_t)bh * SEQ * HD;
  for (int idx = t; idx < NB * HD; idx += 256) {
    int l = idx >> 6, a = idx & 63;
    int s = l * BS + j;
    bf16_t v = (bf16_t)((s < SEQ) ? Qbh[(size_t)s * HD + a] : 0.0f);
    sQ[l][a] = v;
    sQT[a][l] = v;
  }
  if (t < NB) sR2[t] = r2ws[((size_t)bh * NB + t) * BS + j];
  float* Lj = Lws + ((size_t)bh * BS + j) * NB * NB;
  for (int idx = t; idx < NB * NB; idx += 256) sL[idx >> 5][idx & 31] = Lj[idx];
  __syncthreads();

  int wave = t >> 5;
  if (wave < 4) {   // qk_l: 32x32 out, K=64 over a
    int tm = (wave >> 1) << 4, tn = (wave & 1) << 4;
    v8f acc = {};
    acc = wmma_bf16(frag_A(&sQ[0][0], HD, tm, 0),  frag_B_nk(&sT[0][0], HD, 0,  tn), acc);
    acc = wmma_bf16(frag_A(&sQ[0][0], HD, tm, 32), frag_B_nk(&sT[0][0], HD, 32, tn), acc);
    store_D_f32(&sQKl[0][0], NB + 1, tm, tn, acc);
  }
  if (t < NB) {     // per-row normalize of L over k
    int l = t;
    float n2 = 0.0f;
    for (int kk = 0; kk < NB; ++kk) { float v = sL[l][kk]; n2 += v * v; }
    float inv = 1.0f / fmaxf(sqrtf(n2), 1e-12f);
    for (int kk = 0; kk < NB; ++kk) {
      float ls = sL[l][kk] * inv;
      sLS[l][kk] = ls;
      sLeftT[kk][l] = (bf16_t)(ls * ls);
    }
  }
  __syncthreads();

  {  // U[k][a] = sum_l left[l][k] * Q[l][a] : M=32(k), N=64(a), K=32(l)
    bf16_t* Uj = Uws + ((size_t)bh * BS + j) * NB * HD;
    int tm = (wave >> 2) << 4, tn = (wave & 3) << 4;
    v8f acc = {};
    acc = wmma_bf16(frag_A(&sLeftT[0][0], NB, tm, 0), frag_B_nk(&sQT[0][0], NB, 0, tn), acc);
    store_D_bf16(Uj, HD, tm, tn, acc);
  }
  if (t < NB) {     // l2[k] = sum_l left^2
    int kk = t;
    float s2 = 0.0f;
    for (int l = 0; l < NB; ++l) { float lf = (float)sLeftT[kk][l]; s2 += lf * lf; }
    l2ws[((size_t)bh * BS + j) * NB + kk] = s2;
  }
  __syncthreads();

  if (t < NB) {     // d_left + update (tangent projection, /||L||)
    int l = t;
    float n2 = 0.0f;
    for (int kk = 0; kk < NB; ++kk) { float v = sL[l][kk]; n2 += v * v; }
    float nl = sqrtf(n2);
    float invn = (nl > 0.0f) ? (1.0f / nl) : 0.0f;
    float dot = 0.0f;
    for (int kk = 0; kk < NB; ++kk) {
      float ls = sLS[l][kk];
      float raw = (sR2[kk] * ls * ls - sQKl[l][kk]) * INV_N2;
      dot += ls * (2.0f * ls * raw);
    }
    for (int kk = 0; kk < NB; ++kk) {
      float ls = sLS[l][kk];
      float raw = (sR2[kk] * ls * ls - sQKl[l][kk]) * INV_N2;
      float gi = (2.0f * ls * raw - ls * dot) * invn;
      Lj[l * NB + kk] = sL[l][kk] - gi;
    }
  }
}

// ---------------------------------------------------------------------------
// S3: per (bh,k): qk_r = U_k @ K_k^T; d_right + update R
// ---------------------------------------------------------------------------
__global__ __launch_bounds__(256)
void k_step_right_update(const float* __restrict__ Kg, const bf16_t* __restrict__ Uws,
                         const float* __restrict__ l2ws, float* __restrict__ Rws) {
  int gid = blockIdx.x;
  int bh = gid / NB, k = gid % NB;
  __shared__ bf16_t sU[BS][HD];      // [j][a]
  __shared__ bf16_t sK[BS][HD];      // [i][a]
  __shared__ float  sQKr[BS][BS];    // [j][i]
  __shared__ float  sL2[BS];
  int t = threadIdx.x;
  // async copy: U_k rows [j][a], global row stride NB*HD, pure bf16 b128 copy
  const bf16_t* Ubh = Uws + (size_t)bh * BS * NB * HD;
  copy_rows_bf16(&sU[0][0], Ubh + (size_t)k * HD, BS, HD, (size_t)NB * HD);
  const float* Kbh = Kg + (size_t)bh * SEQ * HD;
  for (int idx = t; idx < BS * HD; idx += 256) {
    int i = idx >> 6, a = idx & 63;
    int s = k * BS + i;
    sK[i][a] = (bf16_t)((s < SEQ) ? Kbh[(size_t)s * HD + a] : 0.0f);
  }
  if (t < BS) sL2[t] = l2ws[((size_t)bh * BS + t) * NB + k];
  __syncthreads();
  int wave = t >> 5;
  for (int tile = wave; tile < 16; tile += 8) {
    int tm = (tile >> 2) << 4, tn = (tile & 3) << 4;
    v8f acc = {};
    acc = wmma_bf16(frag_A(&sU[0][0], HD, tm, 0),  frag_B_nk(&sK[0][0], HD, 0,  tn), acc);
    acc = wmma_bf16(frag_A(&sU[0][0], HD, tm, 32), frag_B_nk(&sK[0][0], HD, 32, tn), acc);
    store_D_f32(&sQKr[0][0], BS, tm, tn, acc);
  }
  __syncthreads();
  float* Rk = Rws + ((size_t)bh * NB + k) * BS * BS;
  if (t < BS) {
    int j = t;
    bool mrow = (k == NB - 1);
    float n2 = 0.0f;
    for (int i = 0; i < BS; ++i) {
      float v = Rk[j * BS + i];
      if (mrow && i >= MASK_I0) v = 0.0f;
      n2 += v * v;
    }
    float nr = sqrtf(n2);
    float inv  = 1.0f / fmaxf(nr, 1e-12f);
    float invn = (nr > 0.0f) ? (1.0f / nr) : 0.0f;
    float l2j = sL2[j];
    float dot = 0.0f;
    for (int i = 0; i < BS; ++i) {
      float v = Rk[j * BS + i];
      if (mrow && i >= MASK_I0) v = 0.0f;
      float rs = v * inv;
      float raw = (l2j * rs * rs - sQKr[j][i]) * INV_N2;
      dot += rs * (2.0f * rs * raw);
    }
    for (int i = 0; i < BS; ++i) {
      float v = Rk[j * BS + i];
      if (mrow && i >= MASK_I0) v = 0.0f;
      float rs = v * inv;
      float raw = (l2j * rs * rs - sQKr[j][i]) * INV_N2;
      float gi = (2.0f * rs * raw - rs * dot) * invn;
      Rk[j * BS + i] = v - gi;
    }
  }
}

// ---------------------------------------------------------------------------
// M1: per (bh,k): Y[k,j,a] = normalize(mask(R))^2 @ V_k
// ---------------------------------------------------------------------------
__global__ __launch_bounds__(256)
void k_mul_right(const float* __restrict__ Vg, const float* __restrict__ Rws,
                 bf16_t* __restrict__ Yws) {
  int gid = blockIdx.x;
  int bh = gid / NB, k = gid % NB;
  __shared__ float  sRow[BS][BS + 1];
  __shared__ bf16_t sRight[BS][BS];
  __shared__ bf16_t sVT[HD][BS];       // [a][i]
  int t = threadIdx.x;
  const float* Rk = Rws + ((size_t)bh * NB + k) * BS * BS;
  for (int idx = t; idx < BS * BS; idx += 256) {
    int j = idx >> 6, i = idx & 63;
    float v = Rk[idx];
    if (k == NB - 1 && i >= MASK_I0) v = 0.0f;
    sRow[j][i] = v;
  }
  const float* Vbh = Vg + (size_t)bh * SEQ * HD;
  for (int idx = t; idx < BS * HD; idx += 256) {
    int i = idx >> 6, a = idx & 63;
    int s = k * BS + i;
    sVT[a][i] = (bf16_t)((s < SEQ) ? Vbh[(size_t)s * HD + a] : 0.0f);
  }
  __syncthreads();
  if (t < BS) {
    int j = t;
    float n2 = 0.0f;
    for (int i = 0; i < BS; ++i) { float v = sRow[j][i]; n2 += v * v; }
    float inv = 1.0f / fmaxf(sqrtf(n2), 1e-12f);
    for (int i = 0; i < BS; ++i) {
      float rs = sRow[j][i] * inv;
      sRight[j][i] = (bf16_t)(rs * rs);
    }
  }
  __syncthreads();
  int wave = t >> 5;
  bf16_t* Yk = Yws + ((size_t)bh * NB + k) * BS * HD;
  for (int tile = wave; tile < 16; tile += 8) {
    int tm = (tile >> 2) << 4, tn = (tile & 3) << 4;
    v8f acc = {};
    acc = wmma_bf16(frag_A(&sRight[0][0], BS, tm, 0),  frag_B_nk(&sVT[0][0], BS, 0,  tn), acc);
    acc = wmma_bf16(frag_A(&sRight[0][0], BS, tm, 32), frag_B_nk(&sVT[0][0], BS, 32, tn), acc);
    store_D_bf16(Yk, HD, tm, tn, acc);
  }
}

// ---------------------------------------------------------------------------
// M2: per (bh,j): Z[l,a] = normalize(L_j)^2 @ Y_j ; scatter rows s=l*64+j < SEQ
// ---------------------------------------------------------------------------
__global__ __launch_bounds__(256)
void k_mul_left(const float* __restrict__ Lws, const bf16_t* __restrict__ Yws,
                float* __restrict__ out) {
  int gid = blockIdx.x;
  int bh = gid / BS, j = gid % BS;
  __shared__ float  sL[NB][NB];
  __shared__ bf16_t sLeft[NB][NB];     // [l][k]
  __shared__ bf16_t sYT[HD][NB];       // [a][k]  (transposed for contiguous B)
  int t = threadIdx.x;
  const float* Lj = Lws + ((size_t)bh * BS + j) * NB * NB;
  for (int idx = t; idx < NB * NB; idx += 256) sL[idx >> 5][idx & 31] = Lj[idx];
  const bf16_t* Ybh = Yws + (size_t)bh * NB * BS * HD;
  for (int idx = t; idx < NB * HD; idx += 256) {
    int kk = idx >> 6, a = idx & 63;
    sYT[a][kk] = Ybh[((size_t)kk * BS + j) * HD + a];
  }
  __syncthreads();
  if (t < NB) {
    int l = t;
    float n2 = 0.0f;
    for (int kk = 0; kk < NB; ++kk) { float v = sL[l][kk]; n2 += v * v; }
    float inv = 1.0f / fmaxf(sqrtf(n2), 1e-12f);
    for (int kk = 0; kk < NB; ++kk) {
      float ls = sL[l][kk] * inv;
      sLeft[l][kk] = (bf16_t)(ls * ls);
    }
  }
  __syncthreads();
  int wave = t >> 5;
  int tm = (wave >> 2) << 4, tn = (wave & 3) << 4;   // M=32(l), N=64(a), K=32(k)
  v8f acc = {};
  acc = wmma_bf16(frag_A(&sLeft[0][0], NB, tm, 0), frag_B_nk(&sYT[0][0], NB, 0, tn), acc);
  // scatter with row bound s = l*BS + j < SEQ
  {
    int lane = threadIdx.x & 31;
    int a  = tn + (lane & 15);
    int mb = tm + ((lane >> 4) << 3);
#pragma unroll
    for (int r = 0; r < 8; ++r) {
      int s = (mb + r) * BS + j;
      if (s < SEQ) out[((size_t)bh * SEQ + s) * HD + a] = acc[r];
    }
  }
}

// ---------------------------------------------------------------------------
extern "C" void kernel_launch(void* const* d_in, const int* in_sizes, int n_in,
                              void* d_out, int out_size, void* d_ws, size_t ws_size,
                              hipStream_t stream) {
  (void)in_sizes; (void)n_in; (void)out_size; (void)ws_size;
  const float* Qg = (const float*)d_in[0];
  const float* Kg = (const float*)d_in[1];
  const float* Vg = (const float*)d_in[2];
  const float* L0 = (const float*)d_in[3];
  const float* R0 = (const float*)d_in[4];
  float* out = (float*)d_out;

  char* ws = (char*)d_ws;
  size_t off = 0;
  const size_t Lbytes  = (size_t)BHN * BS * NB * NB * sizeof(float);   // 16 MB
  const size_t Rbytes  = (size_t)BHN * NB * BS * BS * sizeof(float);   // 32 MB
  const size_t Tbytes  = (size_t)BHN * NB * BS * HD * sizeof(bf16_t);  // 16 MB
  const size_t Ubytes  = (size_t)BHN * BS * NB * HD * sizeof(bf16_t);  // 16 MB
  const size_t r2bytes = (size_t)BHN * NB * BS * sizeof(float);
  const size_t l2bytes = (size_t)BHN * BS * NB * sizeof(float);

  float*  Lws  = (float*)(ws + off);  off += Lbytes;
  float*  Rws  = (float*)(ws + off);  off += Rbytes;
  bf16_t* Tws  = (bf16_t*)(ws + off); off += Tbytes;   // reused as Y at the end
  bf16_t* Uws  = (bf16_t*)(ws + off); off += Ubytes;
  float*  r2ws = (float*)(ws + off);  off += r2bytes;
  float*  l2ws = (float*)(ws + off);  off += l2bytes;

  (void)hipMemcpyAsync(Lws, L0, Lbytes, hipMemcpyDeviceToDevice, stream);
  (void)hipMemcpyAsync(Rws, R0, Rbytes, hipMemcpyDeviceToDevice, stream);

  dim3 blk(256);
  for (int step = 0; step < 3; ++step) {
    k_step_right_T<<<BHN * NB, blk, 0, stream>>>(Kg, Rws, Tws, r2ws);
    k_step_left<<<BHN * BS, blk, 0, stream>>>(Qg, Tws, r2ws, Lws, Uws, l2ws);
    k_step_right_update<<<BHN * NB, blk, 0, stream>>>(Kg, Uws, l2ws, Rws);
  }
  bf16_t* Yws = Tws;  // T no longer needed
  k_mul_right<<<BHN * NB, blk, 0, stream>>>(Vg, Rws, Yws);
  k_mul_left<<<BHN * BS, blk, 0, stream>>>(Lws, Yws, out);
}
